// GraphSCNet_10943576670841
// MI455X (gfx1250) — matrix operand order repaired and checked by reading, not compiled, not run on hardware
//
#include <hip/hip_runtime.h>
#include <hip/hip_bf16.h>
#include <math.h>

// ---------------- problem constants ----------------
#define Pp   40000
#define Cc   30000
#define Nn   1024
#define Dd   256
#define Hh   4
#define DHh  64
#define DFFf 512
#define Kl   128
#define Ll   4
#define CAP  1024

typedef __attribute__((ext_vector_type(16))) __bf16 v16bf;
typedef __attribute__((ext_vector_type(8)))  float  v8f;

// ---------------- WMMA helpers (gfx1250 wave32 layouts) ----------------
__device__ __forceinline__ v8f wmma_bf16(v16bf a, v16bf b, v8f c) {
  return __builtin_amdgcn_wmma_f32_16x16x32_bf16(false, a, false, b, (short)0, c,
                                                 false, false);
}

__device__ __forceinline__ v8f zero8() {
  v8f z;
#pragma unroll
  for (int i = 0; i < 8; ++i) z[i] = 0.f;
  return z;
}

// A fragment: 16(M) x 32(K) bf16, row-major source, leading dim ld (elements).
// lane l (l<16): M=l, K=0..15 ; lanes 16..31: M=l-16, K=16..31 (matches frag_bT).
// Per-lane reads are one contiguous 32-byte run -> ds_load_b128 pair.
__device__ __forceinline__ v16bf frag_a(const __bf16* p, int ld) {
  const int lane = threadIdx.x & 31;
  const int m = lane & 15, half = lane >> 4;
  v16bf r;
#pragma unroll
  for (int e = 0; e < 16; ++e) {
    const int k = half * 16 + e;
    r[e] = p[m * ld + k];
  }
  return r;
}

// B fragment (32K x 16N) from a TRANSPOSED source: element (k,n) read from
// src[n*ld + k]. lanes 0..15 hold K=0..15 (col n=lane), lanes 16..31 K=16..31.
// Per-lane reads are one contiguous 32-byte run -> ds_load_b128 pair.
__device__ __forceinline__ v16bf frag_bT(const __bf16* p, int ld) {
  const int lane = threadIdx.x & 31;
  const int n = lane & 15, half = lane >> 4;
  v16bf r;
#pragma unroll
  for (int e = 0; e < 16; ++e) r[e] = p[n * ld + half * 16 + e];
  return r;
}

// ---------------- kNN anchors ----------------
__global__ __launch_bounds__(256) void k_knn(const float* __restrict__ src_points,
                                             const float* __restrict__ scp,
                                             const int* __restrict__ nidx,
                                             int* __restrict__ cand_cnt,
                                             int* __restrict__ cand_id,
                                             float* __restrict__ cand_w) {
  __shared__ float nx[Nn], ny[Nn], nz[Nn];
  for (int i = threadIdx.x; i < Nn; i += 256) {
    const int p = nidx[i];
    nx[i] = src_points[p * 3 + 0];
    ny[i] = src_points[p * 3 + 1];
    nz[i] = src_points[p * 3 + 2];
  }
  __syncthreads();
  const int c = blockIdx.x * 256 + threadIdx.x;
  if (c >= Cc) return;
  const float px = scp[c * 3 + 0], py = scp[c * 3 + 1], pz = scp[c * 3 + 2];
  float bd[6];
  int bi[6];
#pragma unroll
  for (int j = 0; j < 6; ++j) { bd[j] = 1e30f; bi[j] = 0; }
  for (int jn = 0; jn < Nn; ++jn) {
    const float dx = px - nx[jn], dy = py - ny[jn], dz = pz - nz[jn];
    const float d2 = dx * dx + dy * dy + dz * dz;
    if (d2 < bd[5]) {
      int pos = 5;
      while (pos > 0 && d2 < bd[pos - 1]) {
        bd[pos] = bd[pos - 1];
        bi[pos] = bi[pos - 1];
        --pos;
      }
      bd[pos] = d2;
      bi[pos] = jn;
    }
  }
  float w[6], wsum = 0.f;
#pragma unroll
  for (int j = 0; j < 6; ++j) {
    w[j] = __expf(-bd[j] / (2.0f * 0.08f * 0.08f));
    wsum += w[j];
  }
  wsum = fmaxf(wsum, 1e-10f);
#pragma unroll
  for (int j = 0; j < 6; ++j) {
    const float aw = w[j] / wsum;
    const int node = bi[j];
    const int pos = atomicAdd(&cand_cnt[node], 1);
    if (pos < CAP) {
      cand_id[node * CAP + pos] = c;
      cand_w[node * CAP + pos] = aw;
    }
  }
}

// ---------------- per-node top-K selection ----------------
__global__ __launch_bounds__(256) void k_topk(const int* __restrict__ cand_cnt,
                                              const int* __restrict__ cand_id,
                                              const float* __restrict__ cand_w,
                                              int* __restrict__ li,
                                              float* __restrict__ lwv) {
  const int n = blockIdx.x;
  __shared__ float w[CAP];
  __shared__ int id[CAP];
  __shared__ float rw[256];
  __shared__ int rpos[256];
  const int cnt = min(cand_cnt[n], CAP);
  for (int i = threadIdx.x; i < CAP; i += 256) {
    if (i < cnt) {
      w[i] = cand_w[n * CAP + i];
      id[i] = cand_id[n * CAP + i];
    } else {
      w[i] = -1.f;
      id[i] = 0x7fffffff;
    }
  }
  __syncthreads();
  const int nsel = min(cnt, Kl);
  const bool node_ok = cnt > 3;  // sum(lm) > MIN_LOCAL
  for (int k = 0; k < Kl; ++k) {
    float bw = -2.f;
    int bp = -1, bid = 0x7fffffff;
    for (int i = threadIdx.x; i < CAP; i += 256) {
      const float wi = w[i];
      if (wi > bw || (wi == bw && id[i] < bid)) { bw = wi; bp = i; bid = id[i]; }
    }
    rw[threadIdx.x] = bw;
    rpos[threadIdx.x] = bp;
    __syncthreads();
    for (int s = 128; s > 0; s >>= 1) {
      if (threadIdx.x < (unsigned)s) {
        const float wa = rw[threadIdx.x], wb = rw[threadIdx.x + s];
        const int pa = rpos[threadIdx.x], pb = rpos[threadIdx.x + s];
        const int ia = (pa >= 0) ? id[pa] : 0x7fffffff;
        const int ib = (pb >= 0) ? id[pb] : 0x7fffffff;
        if (wb > wa || (wb == wa && ib < ia)) {
          rw[threadIdx.x] = wb;
          rpos[threadIdx.x] = pb;
        }
      }
      __syncthreads();
    }
    if (threadIdx.x == 0) {
      const int p = rpos[0];
      const bool sel = (k < nsel) && (p >= 0);
      li[n * Kl + k] = sel ? id[p] : 0;
      lwv[n * Kl + k] = (sel && node_ok) ? w[p] : 0.f;
      if (p >= 0) w[p] = -1.f;
    }
    __syncthreads();
  }
}

// ---------------- spatial consistency bias (with column mask folded) ----------------
__global__ __launch_bounds__(256) void k_cons(const float* __restrict__ scp,
                                              const float* __restrict__ tcp,
                                              const int* __restrict__ li,
                                              const float* __restrict__ lwv,
                                              float* __restrict__ bias) {
  const int n = blockIdx.x;
  __shared__ float ls[Kl][3], lt[Kl][3];
  __shared__ float colw[Kl];
  for (int i = threadIdx.x; i < Kl; i += 256) {
    const int c = li[n * Kl + i];
    ls[i][0] = scp[c * 3 + 0]; ls[i][1] = scp[c * 3 + 1]; ls[i][2] = scp[c * 3 + 2];
    lt[i][0] = tcp[c * 3 + 0]; lt[i][1] = tcp[c * 3 + 1]; lt[i][2] = tcp[c * 3 + 2];
    colw[i] = lwv[n * Kl + i];
  }
  __syncthreads();
  for (int idx = threadIdx.x; idx < Kl * Kl; idx += 256) {
    const int k = idx >> 7, l2 = idx & 127;
    float b;
    if (colw[l2] > 0.f) {
      const float dx = ls[k][0] - ls[l2][0], dy = ls[k][1] - ls[l2][1],
                  dz = ls[k][2] - ls[l2][2];
      const float dsv = sqrtf(dx * dx + dy * dy + dz * dz);
      const float tx = lt[k][0] - lt[l2][0], ty = lt[k][1] - lt[l2][1],
                  tz = lt[k][2] - lt[l2][2];
      const float dtv = sqrtf(tx * tx + ty * ty + tz * tz);
      const float diff = dsv - dtv;
      const float c1 = fmaxf(1.f - diff * diff / (0.08f * 0.08f), 0.f);
      b = __logf(c1 + 1e-6f);
    } else {
      b = -1e9f;
    }
    bias[(size_t)n * Kl * Kl + idx] = b;
  }
}

// ---------------- input projection ----------------
__global__ __launch_bounds__(256) void k_inproj(const float* __restrict__ scp,
                                                const float* __restrict__ tcp,
                                                const float* __restrict__ w,
                                                const float* __restrict__ b,
                                                float* __restrict__ f) {
  const int idx = blockIdx.x * 256 + threadIdx.x;
  if (idx >= Cc * Dd) return;
  const int c = idx >> 8, d = idx & 255;
  float acc = b[d];
#pragma unroll
  for (int i = 0; i < 3; ++i) {
    acc += scp[c * 3 + i] * w[i * Dd + d];
    acc += tcp[c * 3 + i] * w[(3 + i) * Dd + d];
  }
  f[idx] = acc;
}

// ---------------- generic tiled WMMA GEMM: C = A(MxK) * B(KxN) ----------------
// BM=128, BN=64, BK=32, 256 threads = 8 waves (4x2), wave tile 32x32.
// Double-buffered LDS + global_prefetch of the next-next B tile.
#define BT_LD 40  // padded leading dim of transposed B tile (32 + 8)
template <bool A_F32, bool OUT_BF16, bool ACCUM, bool RELU, bool BIAS>
__global__ __launch_bounds__(256) void k_gemm(const void* __restrict__ Aptr,
                                              const __bf16* __restrict__ B,
                                              const float* __restrict__ bias,
                                              float* __restrict__ Cf,
                                              __bf16* __restrict__ Cb,
                                              int M, int N, int K) {
  __shared__ __bf16 sA[2][128 * 32];
  __shared__ __bf16 sBT[2][64 * BT_LD];
  const int row0 = blockIdx.x * 128, col0 = blockIdx.y * 64;
  const int tid = threadIdx.x;
  const int wave = tid >> 5, lane = tid & 31;
  const int wm = wave >> 1, wn = wave & 1;
  v8f acc[2][2];
#pragma unroll
  for (int i = 0; i < 2; ++i)
#pragma unroll
    for (int j = 0; j < 2; ++j) acc[i][j] = zero8();

  auto load_tiles = [&](int buf, int kk) {
    if constexpr (A_F32) {
      const float* A = (const float*)Aptr;
      for (int i = tid; i < 128 * 32; i += 256) {
        const int r = i >> 5, c = i & 31;
        const int gr = row0 + r;
        const float v = (gr < M) ? A[(size_t)gr * K + kk + c] : 0.f;
        sA[buf][r * 32 + c] = (__bf16)v;
      }
    } else {
      const __bf16* A = (const __bf16*)Aptr;
      for (int i = tid; i < 128 * 32; i += 256) {
        const int r = i >> 5, c = i & 31;
        const int gr = row0 + r;
        sA[buf][r * 32 + c] = (gr < M) ? A[(size_t)gr * K + kk + c] : (__bf16)0.f;
      }
    }
    // B tile staged TRANSPOSED: coalesced global read, sBT[n][k] write.
    for (int i = tid; i < 32 * 64; i += 256) {
      const int r = i >> 6, c = i & 63;  // r = k row, c = n col
      sBT[buf][c * BT_LD + r] = B[(size_t)(kk + r) * N + col0 + c];
    }
  };
  auto compute = [&](int buf) {
#pragma unroll
    for (int mt = 0; mt < 2; ++mt) {
      v16bf af = frag_a(&sA[buf][(wm * 32 + mt * 16) * 32], 32);
#pragma unroll
      for (int nt = 0; nt < 2; ++nt) {
        v16bf bfv = frag_bT(&sBT[buf][(wn * 32 + nt * 16) * BT_LD], BT_LD);
        acc[mt][nt] = wmma_bf16(af, bfv, acc[mt][nt]);
      }
    }
  };

  load_tiles(0, 0);
  __syncthreads();
  int cur = 0;
  for (int kk = 32; kk < K; kk += 32) {
    load_tiles(cur ^ 1, kk);
    if (kk + 32 < K && tid < 32) {
      // speculative prefetch of the B tile after next (one row per lane)
      __builtin_prefetch(&B[(size_t)(kk + 32 + lane) * N + col0], 0, 1);
    }
    compute(cur);
    __syncthreads();
    cur ^= 1;
  }
  compute(cur);

  const int cn = lane & 15, half = lane >> 4;
#pragma unroll
  for (int mt = 0; mt < 2; ++mt)
#pragma unroll
    for (int nt = 0; nt < 2; ++nt) {
      const int colg = col0 + wn * 32 + nt * 16 + cn;
#pragma unroll
      for (int j = 0; j < 8; ++j) {
        const int rowg = row0 + wm * 32 + mt * 16 + j + 8 * half;
        if (rowg < M) {
          float v = acc[mt][nt][j];
          if constexpr (BIAS) v += bias[colg];
          if constexpr (ACCUM) v += Cf[(size_t)rowg * N + colg];
          if constexpr (RELU) v = fmaxf(v, 0.f);
          if constexpr (OUT_BF16)
            Cb[(size_t)rowg * N + colg] = (__bf16)v;
          else
            Cf[(size_t)rowg * N + colg] = v;
        }
      }
    }
}

// ---------------- fused attention per (node, head) ----------------
__global__ __launch_bounds__(256) void k_attn(const __bf16* __restrict__ Q,
                                              const __bf16* __restrict__ Km,
                                              const __bf16* __restrict__ V,
                                              const float* __restrict__ bias,
                                              __bf16* __restrict__ O) {
  __shared__ __bf16 smem[3 * Kl * DHh];  // 48 KB
  __bf16* sQ = smem;                     // [128][64] row-major
  __bf16* sK = smem + Kl * DHh;          // [128][64] row-major
  __bf16* sVT = smem + 2 * Kl * DHh;     // [64][128]  V transposed (d-major)
  __bf16* sS = smem;  // 128x128 bf16 overlays sQ+sK after scores are consumed
  const int n = blockIdx.x, h = blockIdx.y;
  const int tid = threadIdx.x;
  const size_t base = ((size_t)n * Kl) * Dd + h * DHh;
  for (int i = tid; i < Kl * DHh; i += 256) {
    const int r = i >> 6, c = i & 63;
    const size_t g = base + (size_t)r * Dd + c;
    sQ[i] = Q[g];
    sK[i] = Km[g];
    sVT[c * Kl + r] = V[g];
  }
  __syncthreads();
  const int wave = tid >> 5, lane = tid & 31;
  const int cn = lane & 15, half = lane >> 4;
  v8f acc[8];
#pragma unroll
  for (int i = 0; i < 8; ++i) acc[i] = zero8();
#pragma unroll
  for (int kk = 0; kk < DHh; kk += 32) {
    v16bf aq = frag_a(&sQ[(wave * 16) * DHh + kk], DHh);
#pragma unroll
    for (int nt = 0; nt < 8; ++nt) {
      // B = K^T : element (k=d, n=l) read from row-major sK[l][d]
      v16bf bk = frag_bT(&sK[(nt * 16) * DHh + kk], DHh);
      acc[nt] = wmma_bf16(aq, bk, acc[nt]);
    }
  }
  const float scale = 0.125f;  // 1/sqrt(64)
  const float* bp = bias + (size_t)n * Kl * Kl;
  float rsum[8];
#pragma unroll
  for (int j = 0; j < 8; ++j) {
    const int row = wave * 16 + j + 8 * half;
    float m = -3.4e38f;
#pragma unroll
    for (int nt = 0; nt < 8; ++nt) {
      const float s = acc[nt][j] * scale + bp[row * Kl + nt * 16 + cn];
      acc[nt][j] = s;
      m = fmaxf(m, s);
    }
    m = fmaxf(m, __shfl_xor(m, 1));
    m = fmaxf(m, __shfl_xor(m, 2));
    m = fmaxf(m, __shfl_xor(m, 4));
    m = fmaxf(m, __shfl_xor(m, 8));
    float s0 = 0.f;
#pragma unroll
    for (int nt = 0; nt < 8; ++nt) {
      const float e = __expf(acc[nt][j] - m);
      acc[nt][j] = e;
      s0 += e;
    }
    s0 += __shfl_xor(s0, 1);
    s0 += __shfl_xor(s0, 2);
    s0 += __shfl_xor(s0, 4);
    s0 += __shfl_xor(s0, 8);
    rsum[j] = s0;
  }
  __syncthreads();  // everyone done reading sQ/sK
#pragma unroll
  for (int j = 0; j < 8; ++j) {
    const int row = wave * 16 + j + 8 * half;
    const float inv = 1.f / rsum[j];
#pragma unroll
    for (int nt = 0; nt < 8; ++nt)
      sS[row * Kl + nt * 16 + cn] = (__bf16)(acc[nt][j] * inv);
  }
  __syncthreads();
  v8f oa[4];
#pragma unroll
  for (int i = 0; i < 4; ++i) oa[i] = zero8();
#pragma unroll
  for (int kk = 0; kk < Kl; kk += 32) {
    v16bf as = frag_a(&sS[(wave * 16) * Kl + kk], Kl);
#pragma unroll
    for (int nt = 0; nt < 4; ++nt) {
      // B = V : element (k=l, n=d) read from transposed sVT[d][l]
      v16bf bv = frag_bT(&sVT[(nt * 16) * Kl + kk], Kl);
      oa[nt] = wmma_bf16(as, bv, oa[nt]);
    }
  }
#pragma unroll
  for (int nt = 0; nt < 4; ++nt)
#pragma unroll
    for (int j = 0; j < 8; ++j) {
      const int row = wave * 16 + j + 8 * half;
      O[base + (size_t)row * Dd + nt * 16 + cn] = (__bf16)oa[nt][j];
    }
}

// ---------------- row layernorm (D=256), one wave per row ----------------
__global__ __launch_bounds__(256) void k_layernorm(float* __restrict__ X,
                                                   const float* __restrict__ g,
                                                   const float* __restrict__ b,
                                                   int rows) {
  const int wave = threadIdx.x >> 5, lane = threadIdx.x & 31;
  const int row = blockIdx.x * 8 + wave;
  if (row >= rows) return;
  float* x = X + (size_t)row * Dd;
  float v[8], m = 0.f;
#pragma unroll
  for (int i = 0; i < 8; ++i) {
    v[i] = x[i * 32 + lane];
    m += v[i];
  }
#pragma unroll
  for (int s = 1; s < 32; s <<= 1) m += __shfl_xor(m, s);
  m *= (1.f / 256.f);
  float var = 0.f;
#pragma unroll
  for (int i = 0; i < 8; ++i) {
    const float d = v[i] - m;
    var += d * d;
  }
#pragma unroll
  for (int s = 1; s < 32; s <<= 1) var += __shfl_xor(var, s);
  var *= (1.f / 256.f);
  const float inv = rsqrtf(var + 1e-5f);
#pragma unroll
  for (int i = 0; i < 8; ++i) {
    const int col = i * 32 + lane;
    x[col] = (v[i] - m) * inv * g[col] + b[col];
  }
}

// ---------------- gather / scatter / finalize ----------------
__global__ __launch_bounds__(256) void k_gatherX(const float* __restrict__ f,
                                                 const int* __restrict__ li,
                                                 float* __restrict__ X) {
  const size_t idx = (size_t)blockIdx.x * 256 + threadIdx.x;
  if (idx >= (size_t)Nn * Kl * Dd) return;
  const int rk = (int)(idx >> 8);
  const int d = (int)(idx & 255);
  X[idx] = f[(size_t)li[rk] * Dd + d];
}

__global__ __launch_bounds__(256) void k_scatter(const float* __restrict__ X,
                                                 const int* __restrict__ li,
                                                 const float* __restrict__ lwv,
                                                 float* __restrict__ num,
                                                 float* __restrict__ den) {
  const int gid = blockIdx.x * 256 + threadIdx.x;
  const int rk = gid >> 7;
  const int t = gid & 127;
  if (rk >= Nn * Kl) return;
  const float w = lwv[rk];
  if (w <= 0.f) return;
  const int c = li[rk];
  const float* x = X + (size_t)rk * Dd;
  atomicAdd(&num[(size_t)c * Dd + 2 * t + 0], w * x[2 * t + 0]);
  atomicAdd(&num[(size_t)c * Dd + 2 * t + 1], w * x[2 * t + 1]);
  if (t == 0) atomicAdd(&den[c], w);
}

__global__ __launch_bounds__(256) void k_finalize(const float* __restrict__ num,
                                                  const float* __restrict__ den,
                                                  float* __restrict__ f,
                                                  float* __restrict__ masks) {
  const int idx = blockIdx.x * 256 + threadIdx.x;
  if (idx >= Cc * Dd) return;
  const int c = idx >> 8;
  const float d = den[c];
  f[idx] = num[idx] / fmaxf(d, 1e-6f);
  if (masks != nullptr && (idx & 255) == 0) masks[c] = (d > 0.f) ? 1.f : 0.f;
}

// ---------------- row L2 normalize ----------------
__global__ __launch_bounds__(256) void k_rownorm(const float* __restrict__ f2,
                                                 float* __restrict__ out, int rows) {
  const int wave = threadIdx.x >> 5, lane = threadIdx.x & 31;
  const int row = blockIdx.x * 8 + wave;
  if (row >= rows) return;
  const float* x = f2 + (size_t)row * Dd;
  float v[8], ss = 0.f;
#pragma unroll
  for (int i = 0; i < 8; ++i) {
    v[i] = x[i * 32 + lane];
    ss += v[i] * v[i];
  }
#pragma unroll
  for (int s = 1; s < 32; s <<= 1) ss += __shfl_xor(ss, s);
  const float inv = 1.f / fmaxf(sqrtf(ss), 1e-12f);
  float* o = out + (size_t)row * Dd;
#pragma unroll
  for (int i = 0; i < 8; ++i) o[i * 32 + lane] = v[i] * inv;
}

// ---------------- GroupNorm stats / apply (+ leaky relu) ----------------
__global__ __launch_bounds__(256) void k_gnstat(const float* __restrict__ x,
                                                float* __restrict__ stats,
                                                int total, int ch, int chPerG) {
  __shared__ float ls[16];
  if (threadIdx.x < 16) ls[threadIdx.x] = 0.f;
  __syncthreads();
  const int idx = blockIdx.x * 256 + threadIdx.x;
  if (idx < total) {
    const float v = x[idx];
    const int g = (idx % ch) / chPerG;
    atomicAdd(&ls[g * 2 + 0], v);
    atomicAdd(&ls[g * 2 + 1], v * v);
  }
  __syncthreads();
  if (threadIdx.x < 16) atomicAdd(&stats[threadIdx.x], ls[threadIdx.x]);
}

__global__ __launch_bounds__(256) void k_gnapply(float* __restrict__ x,
                                                 const float* __restrict__ stats,
                                                 const float* __restrict__ g,
                                                 const float* __restrict__ b,
                                                 int total, int ch, int chPerG,
                                                 float cntInv) {
  const int idx = blockIdx.x * 256 + threadIdx.x;
  if (idx >= total) return;
  const int col = idx % ch;
  const int grp = col / chPerG;
  const float m = stats[grp * 2 + 0] * cntInv;
  const float var = stats[grp * 2 + 1] * cntInv - m * m;
  float y = (x[idx] - m) * rsqrtf(var + 1e-5f) * g[col] + b[col];
  x[idx] = (y > 0.f) ? y : 0.01f * y;  // leaky_relu(0.01)
}

// ---------------- classifier head ----------------
__global__ __launch_bounds__(256) void k_cls3(const float* __restrict__ h,
                                              const float* __restrict__ w,
                                              const float* __restrict__ b,
                                              float* __restrict__ logits,
                                              float* __restrict__ scores) {
  const int c = blockIdx.x * 256 + threadIdx.x;
  if (c >= Cc) return;
  float a = b[0];
#pragma unroll 8
  for (int i = 0; i < 64; ++i) a += h[(size_t)c * 64 + i] * w[i];
  logits[c] = a;
  scores[c] = 1.f / (1.f + __expf(-a));
}

// ---------------- local feature consistency (Gram via WMMA) ----------------
__global__ __launch_bounds__(256) void k_localfc(const float* __restrict__ cfn,
                                                 const int* __restrict__ li,
                                                 const float* __restrict__ sigma,
                                                 float* __restrict__ out) {
  __shared__ __bf16 sL[Kl * Dd];  // 64 KB
  const int n = blockIdx.x;
  for (int i = threadIdx.x; i < Kl * Dd; i += 256) {
    const int r = i >> 8, d = i & 255;
    const int c = li[n * Kl + r];
    sL[i] = (__bf16)cfn[(size_t)c * Dd + d];
  }
  __syncthreads();
  const int wave = threadIdx.x >> 5, lane = threadIdx.x & 31;
  const int cn = lane & 15, half = lane >> 4;
  v8f acc[8];
#pragma unroll
  for (int i = 0; i < 8; ++i) acc[i] = zero8();
  for (int kk = 0; kk < Dd; kk += 32) {
    v16bf a = frag_a(&sL[(wave * 16) * Dd + kk], Dd);
#pragma unroll
    for (int nt = 0; nt < 8; ++nt) {
      // B = L^T : element (k=d, n=l) read from row-major sL[l][d]
      v16bf bt = frag_bT(&sL[(nt * 16) * Dd + kk], Dd);
      acc[nt] = wmma_bf16(a, bt, acc[nt]);
    }
  }
  const float s2 = sigma[0] * sigma[0];
#pragma unroll
  for (int nt = 0; nt < 8; ++nt)
#pragma unroll
    for (int j = 0; j < 8; ++j) {
      const int row = wave * 16 + j + 8 * half;
      const int col = nt * 16 + cn;
      const float gv = acc[nt][j];
      const float aff2 = fmaxf(2.f - 2.f * gv, 0.f);
      out[(size_t)n * Kl * Kl + row * Kl + col] = fmaxf(1.f - aff2 / s2, 0.f);
    }
}

// ---------------- f32 -> bf16 conversion ----------------
__global__ __launch_bounds__(256) void k_f2b(const float* __restrict__ s,
                                             __bf16* __restrict__ d, int n) {
  const int i = blockIdx.x * 256 + threadIdx.x;
  if (i < n) d[i] = (__bf16)s[i];
}

// ---------------- host orchestration ----------------
extern "C" void kernel_launch(void* const* d_in, const int* in_sizes, int n_in,
                              void* d_out, int out_size, void* d_ws, size_t ws_size,
                              hipStream_t stream) {
  const float* src_points = (const float*)d_in[0];
  const float* scp = (const float*)d_in[2];
  const float* tcp = (const float*)d_in[3];
  const int* nidx = (const int*)d_in[4];
  const float* in_w = (const float*)d_in[5];
  const float* in_b = (const float*)d_in[6];
  const float* wq = (const float*)d_in[7];
  const float* wk = (const float*)d_in[8];
  const float* wv = (const float*)d_in[9];
  const float* wo = (const float*)d_in[10];
  const float* ln1_g = (const float*)d_in[11];
  const float* ln1_b = (const float*)d_in[12];
  const float* ffn_w1 = (const float*)d_in[13];
  const float* ffn_b1 = (const float*)d_in[14];
  const float* ffn_w2 = (const float*)d_in[15];
  const float* ffn_b2 = (const float*)d_in[16];
  const float* ln2_g = (const float*)d_in[17];
  const float* ln2_b = (const float*)d_in[18];
  const float* out_w = (const float*)d_in[19];
  const float* out_b = (const float*)d_in[20];
  const float* cls_w1 = (const float*)d_in[21];
  const float* cls_b1 = (const float*)d_in[22];
  const float* gn1_g = (const float*)d_in[23];
  const float* gn1_b = (const float*)d_in[24];
  const float* cls_w2 = (const float*)d_in[25];
  const float* cls_b2 = (const float*)d_in[26];
  const float* gn2_g = (const float*)d_in[27];
  const float* gn2_b = (const float*)d_in[28];
  const float* cls_w3 = (const float*)d_in[29];
  const float* cls_b3 = (const float*)d_in[30];
  const float* sigma_f = (const float*)d_in[31];

  // output slices
  float* out_logits = (float*)d_out;
  float* out_scores = out_logits + Cc;
  float* out_cfn = out_scores + Cc;
  float* out_localfc = out_cfn + (size_t)Cc * Dd;
  float* out_masks = out_localfc + (size_t)Nn * Kl * Kl;

  // workspace carve
  char* p = (char*)d_ws;
  auto alloc = [&](size_t bytes) {
    char* r = p;
    p += (bytes + 255) & ~(size_t)255;
    return r;
  };
  int* cand_cnt = (int*)alloc((size_t)Nn * 4);
  int* cand_id = (int*)alloc((size_t)Nn * CAP * 4);
  float* cand_w = (float*)alloc((size_t)Nn * CAP * 4);
  int* li = (int*)alloc((size_t)Nn * Kl * 4);
  float* lwv = (float*)alloc((size_t)Nn * Kl * 4);
  float* bias = (float*)alloc((size_t)Nn * Kl * Kl * 4);
  float* fbuf = (float*)alloc((size_t)Cc * Dd * 4);
  float* f2 = (float*)alloc((size_t)Cc * Dd * 4);
  float* X = (float*)alloc((size_t)Nn * Kl * Dd * 4);
  __bf16* Qb = (__bf16*)alloc((size_t)Nn * Kl * Dd * 2);
  __bf16* Kb = (__bf16*)alloc((size_t)Nn * Kl * Dd * 2);
  __bf16* Vb = (__bf16*)alloc((size_t)Nn * Kl * Dd * 2);
  __bf16* Ob = (__bf16*)alloc((size_t)Nn * Kl * Dd * 2);
  __bf16* H1b = Qb;  // FFN hidden (128 MB) reuses dead Q+K region
  float* num = (float*)alloc((size_t)Cc * Dd * 4);
  float* den = (float*)alloc((size_t)Cc * 4);
  __bf16* wq_b = (__bf16*)alloc((size_t)Ll * Dd * Dd * 2);
  __bf16* wk_b = (__bf16*)alloc((size_t)Ll * Dd * Dd * 2);
  __bf16* wv_b = (__bf16*)alloc((size_t)Ll * Dd * Dd * 2);
  __bf16* wo_b = (__bf16*)alloc((size_t)Ll * Dd * Dd * 2);
  __bf16* w1_b = (__bf16*)alloc((size_t)Ll * Dd * DFFf * 2);
  __bf16* w2_b = (__bf16*)alloc((size_t)Ll * DFFf * Dd * 2);
  __bf16* outw_b = (__bf16*)alloc((size_t)Dd * Dd * 2);
  __bf16* cls1_b = (__bf16*)alloc((size_t)Dd * 128 * 2);
  __bf16* cls2_b = (__bf16*)alloc((size_t)128 * 64 * 2);
  float* h1c = (float*)alloc((size_t)Cc * 128 * 4);
  float* h2c = (float*)alloc((size_t)Cc * 64 * 4);
  float* gnstats = (float*)alloc(32 * 4);
  (void)ws_size;
  (void)in_sizes;
  (void)n_in;
  (void)out_size;

  const int NK = Nn * Kl;  // 131072 rows

  // ---- weight conversions to bf16 ----
  auto f2b = [&](const float* s, __bf16* d, int n) {
    k_f2b<<<(n + 255) / 256, 256, 0, stream>>>(s, d, n);
  };
  f2b(wq, wq_b, Ll * Dd * Dd);
  f2b(wk, wk_b, Ll * Dd * Dd);
  f2b(wv, wv_b, Ll * Dd * Dd);
  f2b(wo, wo_b, Ll * Dd * Dd);
  f2b(ffn_w1, w1_b, Ll * Dd * DFFf);
  f2b(ffn_w2, w2_b, Ll * DFFf * Dd);
  f2b(out_w, outw_b, Dd * Dd);
  f2b(cls_w1, cls1_b, Dd * 128);
  f2b(cls_w2, cls2_b, 128 * 64);

  // ---- graph construction ----
  hipMemsetAsync(cand_cnt, 0, (size_t)Nn * 4, stream);
  k_knn<<<(Cc + 255) / 256, 256, 0, stream>>>(src_points, scp, nidx, cand_cnt,
                                              cand_id, cand_w);
  k_topk<<<Nn, 256, 0, stream>>>(cand_cnt, cand_id, cand_w, li, lwv);
  k_cons<<<Nn, 256, 0, stream>>>(scp, tcp, li, lwv, bias);
  k_inproj<<<(Cc * Dd) / 256, 256, 0, stream>>>(scp, tcp, in_w, in_b, fbuf);

  for (int b = 0; b < 2; ++b) {
    k_gatherX<<<(NK * Dd) / 256, 256, 0, stream>>>(fbuf, li, X);
    for (int j = 0; j < 2; ++j) {
      const int l = b * 2 + j;
      // QKV projections: [NK,256] x [256,256] -> bf16
      dim3 g1(NK / 128, Dd / 64);
      k_gemm<true, true, false, false, false><<<g1, 256, 0, stream>>>(
          X, wq_b + (size_t)l * Dd * Dd, nullptr, nullptr, Qb, NK, Dd, Dd);
      k_gemm<true, true, false, false, false><<<g1, 256, 0, stream>>>(
          X, wk_b + (size_t)l * Dd * Dd, nullptr, nullptr, Kb, NK, Dd, Dd);
      k_gemm<true, true, false, false, false><<<g1, 256, 0, stream>>>(
          X, wv_b + (size_t)l * Dd * Dd, nullptr, nullptr, Vb, NK, Dd, Dd);
      // attention
      dim3 ga(Nn, Hh);
      k_attn<<<ga, 256, 0, stream>>>(Qb, Kb, Vb, bias, Ob);
      // X += O @ Wo ; layernorm1
      k_gemm<false, false, true, false, false><<<g1, 256, 0, stream>>>(
          Ob, wo_b + (size_t)l * Dd * Dd, nullptr, X, nullptr, NK, Dd, Dd);
      k_layernorm<<<NK / 8, 256, 0, stream>>>(X, ln1_g + l * Dd, ln1_b + l * Dd, NK);
      // FFN: H1 = relu(X@W1+b1) ; X += H1@W2+b2 ; layernorm2
      dim3 g2(NK / 128, DFFf / 64);
      k_gemm<true, true, false, true, true><<<g2, 256, 0, stream>>>(
          X, w1_b + (size_t)l * Dd * DFFf, ffn_b1 + l * DFFf, nullptr, H1b, NK,
          DFFf, Dd);
      k_gemm<false, false, true, false, true><<<g1, 256, 0, stream>>>(
          H1b, w2_b + (size_t)l * DFFf * Dd, ffn_b2 + l * Dd, X, nullptr, NK, Dd,
          DFFf);
      k_layernorm<<<NK / 8, 256, 0, stream>>>(X, ln2_g + l * Dd, ln2_b + l * Dd, NK);
    }
    // weighted scatter back to per-correspondence features
    hipMemsetAsync(num, 0, (size_t)Cc * Dd * 4, stream);
    hipMemsetAsync(den, 0, (size_t)Cc * 4, stream);
    k_scatter<<<(NK * 128) / 256, 256, 0, stream>>>(X, li, lwv, num, den);
    k_finalize<<<(Cc * Dd) / 256, 256, 0, stream>>>(num, den, fbuf,
                                                    (b == 1) ? out_masks : nullptr);
  }

  // output projection + normalized features
  {
    dim3 g((Cc + 127) / 128, Dd / 64);
    k_gemm<true, false, false, false, true><<<g, 256, 0, stream>>>(
        fbuf, outw_b, out_b, f2, nullptr, Cc, Dd, Dd);
  }
  k_rownorm<<<(Cc + 7) / 8, 256, 0, stream>>>(f2, out_cfn, Cc);

  // classifier
  hipMemsetAsync(gnstats, 0, 32 * 4, stream);
  {
    dim3 g((Cc + 127) / 128, 128 / 64);
    k_gemm<true, false, false, false, true><<<g, 256, 0, stream>>>(
        f2, cls1_b, cls_b1, h1c, nullptr, Cc, 128, Dd);
  }
  k_gnstat<<<(Cc * 128 + 255) / 256, 256, 0, stream>>>(h1c, gnstats, Cc * 128, 128, 16);
  k_gnapply<<<(Cc * 128 + 255) / 256, 256, 0, stream>>>(
      h1c, gnstats, gn1_g, gn1_b, Cc * 128, 128, 16, 1.f / ((float)Cc * 16.f));
  {
    dim3 g((Cc + 127) / 128, 1);
    k_gemm<true, false, false, false, true><<<g, 256, 0, stream>>>(
        h1c, cls2_b, cls_b2, h2c, nullptr, Cc, 64, 128);
  }
  k_gnstat<<<(Cc * 64 + 255) / 256, 256, 0, stream>>>(h2c, gnstats + 16, Cc * 64, 64, 8);
  k_gnapply<<<(Cc * 64 + 255) / 256, 256, 0, stream>>>(
      h2c, gnstats + 16, gn2_g, gn2_b, Cc * 64, 64, 8, 1.f / ((float)Cc * 8.f));
  k_cls3<<<(Cc + 255) / 256, 256, 0, stream>>>(h2c, cls_w3, cls_b3, out_logits,
                                               out_scores);

  // local feature-consistency matrix
  k_localfc<<<Nn, 256, 0, stream>>>(out_cfn, li, sigma_f, out_localfc);
}
